// iRMB_21680994910416
// MI455X (gfx1250) — matrix operand order repaired and checked by reading, not compiled, not run on hardware
//
#include <hip/hip_runtime.h>
#include <hip/hip_bf16.h>
#include <math.h>

typedef __attribute__((ext_vector_type(16))) _Float16     v16h;
typedef __attribute__((ext_vector_type(8)))  float        v8f;
typedef __attribute__((ext_vector_type(4)))  unsigned int v4u;

#define DIM  256
#define NH   4
#define HC   64
#define NTOK 49
#define MPAD 64
#define QK_SCALE 0.125f   // HC^-0.5

// ---------------- WMMA helpers ----------------

__device__ __forceinline__ v8f wmma32(v16h a, v16h b, v8f c) {
  // D = A(16x32 f16) x B(32x16 f16) + C(16x16 f32)
  return __builtin_amdgcn_wmma_f32_16x16x32_f16(
      /*neg_a=*/false, a, /*neg_b=*/false, b,
      /*c_mod=*/(short)0, c, /*reuse_a=*/false, /*reuse_b=*/false);
}

// A-matrix 16x32 f16 fragment from row-major [row][k] storage.
// ISA layout: lanes 0-15 (M=lane): K {0..7,16..23}; lanes 16-31 (M=lane-16): K {8..15,24..31}
// Two contiguous 16B chunks per lane -> 2 x b128 loads.
__device__ __forceinline__ v16h load_a_frag(const _Float16* base, int row0, int stride, int k0) {
  const int lane = threadIdx.x & 31;
  union { v16h v; v4u q[2]; } r;
  const _Float16* p = base + (row0 + (lane & 15)) * stride + k0 + ((lane & 16) ? 8 : 0);
  r.q[0] = *(const v4u*)p;
  r.q[1] = *(const v4u*)(p + 16);
  return r.v;
}

// B-matrix 32x16 f16 fragment. Storage indexed [N][K] row-major (transposed weight /
// key rows / vT). Lanes 0-15 (N=lane): K {0..7,16..23}; lanes 16-31: K {8..15,24..31}.
// Two 16B chunks per lane (K offsets 0 and 16) -> 2 x b128 loads.
__device__ __forceinline__ v16h load_b_frag(const _Float16* base, int n0, int stride, int k0) {
  const int lane = threadIdx.x & 31;
  union { v16h v; v4u q[2]; } r;
  const _Float16* p = base + (n0 + (lane & 15)) * stride + k0 + ((lane & 16) ? 8 : 0);
  r.q[0] = *(const v4u*)p;
  r.q[1] = *(const v4u*)(p + 16);
  return r.v;
}

// ---------------- Kernel 0: weight/bias standardization ----------------
__global__ void prep_kernel(const float* w0, const float* b0, const float* w1, const float* b1,
                            const float* w2, const float* b2, const float* w3, const float* b3,
                            const float* w4, const float* b4,
                            _Float16* __restrict__ wt, float* __restrict__ bstd) {
  const float* W[5] = {w0, w1, w2, w3, w4};
  const float* B[5] = {b0, b1, b2, b3, b4};
  int t = blockIdx.x * blockDim.x + threadIdx.x;
  if (t < 5 * 1024) {                 // 5 mats x 256 rows x 4 heads
    int mat = t >> 10, rr = t & 1023, k = rr >> 2, h = rr & 3;
    const float* w = W[mat] + k * DIM + h * HC;
    float s = 0.f, ss = 0.f;
    for (int c = 0; c < HC; ++c) { float v = w[c]; s += v; ss += v * v; }
    float m = s * (1.0f / HC);
    float var = ss * (1.0f / HC) - m * m;
    float inv = 1.0f / (sqrtf(var) * 16.0f);     // 16 = sqrt(DIM)
    _Float16* o = wt + mat * (DIM * DIM);
    for (int c = 0; c < HC; ++c)
      o[(h * HC + c) * DIM + k] = (_Float16)((w[c] - m) * inv);   // Wt[n][k]
  } else if (t < 5 * 1024 + 20) {     // 5 biases x 4 heads
    int bt = t - 5120, mat = bt >> 2, h = bt & 3;
    const float* b = B[mat] + h * HC;
    float s = 0.f, ss = 0.f;
    for (int c = 0; c < HC; ++c) { float v = b[c]; s += v; ss += v * v; }
    float m = s * (1.0f / HC);
    float var = ss * (1.0f / HC) - m * m;
    float inv = 1.0f / (sqrtf(var) * 16.0f);
    for (int c = 0; c < HC; ++c) bstd[mat * DIM + h * HC + c] = (b[c] - m) * inv;
  }
}

// ---------------- Kernel 1: fully fused iRMB block, one batch elem / workgroup ----------------
__global__ __launch_bounds__(256) void irmb_kernel(const float* __restrict__ x,
                                                   const _Float16* __restrict__ wt,
                                                   const float* __restrict__ bstd,
                                                   float* __restrict__ out) {
  extern __shared__ char smem[];
  _Float16* s_x16 = (_Float16*)(smem);             // [64][256] x f16, later x1 f16
  _Float16* s_q16 = (_Float16*)(smem + 32768);     // [64][256] q f16, later P [4][64][64]
  _Float16* s_k16 = (_Float16*)(smem + 65536);     // [64][256] k*scale f16, later x4 f16
  _Float16* s_vT  = (_Float16*)(smem + 98304);     // [256][64] v transposed f16
  float*    s_res = (float*)(smem + 131072);       // [64][256] f32: x -> v -> x1 -> x4
  float*    s_S   = (float*)(smem + 196608);       // [4][64][64] scores f32

  const int tid  = threadIdx.x;
  const int lane = tid & 31;
  const int wave = tid >> 5;
  const int b    = blockIdx.x;
  const float* xb = x + (size_t)b * NTOK * DIM;

  // ---- phase 1: async-stage x into LDS (ASYNCcnt path), zero pads, convert to f16 ----
  {
    const uint32_t lds_base = (uint32_t)(uintptr_t)(void*)s_res;  // flat->LDS: low 32 bits
    const char* gbase = (const char*)xb;
    // 49*256 f32 = 3136 chunks of 16B (block base is 16B aligned: 50176 = 16*3136)
    for (int c = tid; c < (NTOK * DIM) / 4; c += 256) {
      uint32_t loff  = lds_base + (uint32_t)(c * 16);
      uint64_t gaddr = (uint64_t)(uintptr_t)(gbase + (size_t)c * 16);
      asm volatile("global_load_async_to_lds_b128 %0, %1, off"
                   :: "v"(loff), "v"(gaddr) : "memory");
    }
    for (int idx = NTOK * DIM + tid; idx < MPAD * DIM; idx += 256) s_res[idx] = 0.0f;
    asm volatile("s_wait_asynccnt 0x0" ::: "memory");
    __syncthreads();
    for (int idx = tid; idx < MPAD * DIM; idx += 256) s_x16[idx] = (_Float16)s_res[idx];
  }
  __syncthreads();

  // ---- phase 2: q,k,v = x @ Wstd + b  (B frags reused across 4 M-tiles) ----
  for (int task = wave; task < 48; task += 8) {
    int mat = task / 16, nt = task % 16;
    const _Float16* wtm = wt + mat * (DIM * DIM);
    v16h Bf[8];
#pragma unroll
    for (int s = 0; s < 8; ++s) Bf[s] = load_b_frag(wtm, nt * 16, DIM, s * 32);
    float bv = bstd[mat * DIM + nt * 16 + (lane & 15)];
    for (int mt = 0; mt < 4; ++mt) {
      v8f acc = {};
#pragma unroll
      for (int s = 0; s < 8; ++s) {
        v16h a = load_a_frag(s_x16, mt * 16, DIM, s * 32);
        acc = wmma32(a, Bf[s], acc);
      }
#pragma unroll
      for (int r = 0; r < 8; ++r) {
        int row = mt * 16 + ((lane & 16) ? r + 8 : r);
        int col = nt * 16 + (lane & 15);
        float val = acc[r] + bv;
        if (mat == 0) {
          s_q16[row * DIM + col] = (_Float16)val;
        } else if (mat == 1) {
          s_k16[row * DIM + col] = (_Float16)(val * QK_SCALE);
        } else {
          float nv = 0.1f * val + 0.95f * s_res[row * DIM + col];   // v = v*0.1 + x*0.95
          s_res[row * DIM + col] = nv;
          s_vT[col * MPAD + row] = (_Float16)nv;
        }
      }
    }
  }
  __syncthreads();

  // ---- phase 3: S = q @ (k*scale)^T per head ----
  for (int task = wave; task < 64; task += 8) {
    int head = task >> 4, rem = task & 15, mt = rem >> 2, nt = rem & 3;
    v8f acc = {};
#pragma unroll
    for (int s = 0; s < 2; ++s) {
      v16h a  = load_a_frag(s_q16, mt * 16, DIM, head * HC + s * 32);
      v16h bb = load_b_frag(s_k16, nt * 16, DIM, head * HC + s * 32);  // [token][chan] = [N][K]
      acc = wmma32(a, bb, acc);
    }
    float* Sp = s_S + head * 4096;
#pragma unroll
    for (int r = 0; r < 8; ++r) {
      int row = mt * 16 + ((lane & 16) ? r + 8 : r);
      int col = nt * 16 + (lane & 15);
      Sp[row * 64 + col] = acc[r];
    }
  }
  __syncthreads();

  // ---- phase 4: masked softmax rows -> P f16 (reuse s_q16) ----
  {
    _Float16* P = s_q16;
    int t = tid;                    // exactly 256 rows: 4 heads x 64 padded queries
    int head = t >> 6, q = t & 63;
    const float* Sr = s_S + head * 4096 + q * 64;
    _Float16* Pr = P + head * 4096 + q * 64;
    if (q < NTOK) {
      float mx = -1e30f;
      for (int j = 0; j < NTOK; ++j) mx = fmaxf(mx, Sr[j]);
      float sum = 0.f;
      for (int j = 0; j < NTOK; ++j) sum += __expf(Sr[j] - mx);
      float rinv = 1.0f / sum;
      for (int j = 0; j < NTOK; ++j) Pr[j] = (_Float16)(__expf(Sr[j] - mx) * rinv);
      for (int j = NTOK; j < 64; ++j) Pr[j] = (_Float16)0.0f;      // mask padded keys
    } else {
      for (int j = 0; j < 64; ++j) Pr[j] = (_Float16)0.0f;
    }
  }
  __syncthreads();

  // ---- phase 5: x_spa = P @ v ; x1 = 0.1*x_spa + 0.95*v ----
  for (int task = wave; task < 64; task += 8) {
    int head = task >> 4, rem = task & 15, mt = rem >> 2, ct = rem & 3;
    v8f acc = {};
#pragma unroll
    for (int s = 0; s < 2; ++s) {
      v16h a  = load_a_frag(s_q16 + head * 4096, mt * 16, 64, s * 32);      // P rows
      v16h bb = load_b_frag(s_vT, head * HC + ct * 16, MPAD, s * 32);       // vT[chan][tok]=[N][K]
      acc = wmma32(a, bb, acc);
    }
#pragma unroll
    for (int r = 0; r < 8; ++r) {
      int row = mt * 16 + ((lane & 16) ? r + 8 : r);
      int col = head * HC + ct * 16 + (lane & 15);
      float x1 = 0.1f * acc[r] + 0.95f * s_res[row * DIM + col];
      s_res[row * DIM + col] = x1;
      s_x16[row * DIM + col] = (_Float16)x1;
    }
  }
  __syncthreads();

  // ---- phase 6: FFN-in GEMM + residual + sigmoid branch ----
  for (int task = wave; task < 16; task += 8) {
    const _Float16* wtm = wt + 3 * (DIM * DIM);
    v16h Bf[8];
#pragma unroll
    for (int s = 0; s < 8; ++s) Bf[s] = load_b_frag(wtm, task * 16, DIM, s * 32);
    float bv = bstd[3 * DIM + task * 16 + (lane & 15)];
    for (int mt = 0; mt < 4; ++mt) {
      v8f acc = {};
#pragma unroll
      for (int s = 0; s < 8; ++s) {
        v16h a = load_a_frag(s_x16, mt * 16, DIM, s * 32);
        acc = wmma32(a, Bf[s], acc);
      }
#pragma unroll
      for (int r = 0; r < 8; ++r) {
        int row = mt * 16 + ((lane & 16) ? r + 8 : r);
        int col = task * 16 + (lane & 15);
        float val = acc[r] + bv;
        float x1  = s_res[row * DIM + col];
        float x3  = 0.1f * val + 0.95f * x1;
        float sg  = 1.0f / (1.0f + __expf(-x3));
        float x4  = (sg - 0.5f) * 0.1f + 0.95f * x3;
        s_res[row * DIM + col] = x4;
        s_k16[row * DIM + col] = (_Float16)x4;
      }
    }
  }
  __syncthreads();

  // ---- phase 7: FFN-out GEMM + final residual + store ----
  for (int task = wave; task < 16; task += 8) {
    const _Float16* wtm = wt + 4 * (DIM * DIM);
    v16h Bf[8];
#pragma unroll
    for (int s = 0; s < 8; ++s) Bf[s] = load_b_frag(wtm, task * 16, DIM, s * 32);
    float bv = bstd[4 * DIM + task * 16 + (lane & 15)];
    for (int mt = 0; mt < 4; ++mt) {
      v8f acc = {};
#pragma unroll
      for (int s = 0; s < 8; ++s) {
        v16h a = load_a_frag(s_k16, mt * 16, DIM, s * 32);
        acc = wmma32(a, Bf[s], acc);
      }
#pragma unroll
      for (int r = 0; r < 8; ++r) {
        int row = mt * 16 + ((lane & 16) ? r + 8 : r);
        int col = task * 16 + (lane & 15);
        float val = acc[r] + bv;
        float o   = 0.1f * val + 0.95f * s_res[row * DIM + col];
        if (row < NTOK)
          out[(size_t)b * NTOK * DIM + row * DIM + col] = o;
      }
    }
  }
}

// ---------------- launch ----------------
extern "C" void kernel_launch(void* const* d_in, const int* in_sizes, int n_in,
                              void* d_out, int out_size, void* d_ws, size_t ws_size,
                              hipStream_t stream) {
  (void)in_sizes; (void)n_in; (void)out_size; (void)ws_size;
  const float* x = (const float*)d_in[0];
  _Float16* wt   = (_Float16*)d_ws;                                  // 5 x 256x256 f16
  float*    bstd = (float*)((char*)d_ws + 5 * DIM * DIM * sizeof(_Float16));

  prep_kernel<<<21, 256, 0, stream>>>(
      (const float*)d_in[1], (const float*)d_in[2],
      (const float*)d_in[3], (const float*)d_in[4],
      (const float*)d_in[5], (const float*)d_in[6],
      (const float*)d_in[7], (const float*)d_in[8],
      (const float*)d_in[9], (const float*)d_in[10],
      wt, bstd);

  hipFuncSetAttribute((const void*)irmb_kernel,
                      hipFuncAttributeMaxDynamicSharedMemorySize, 262144);
  irmb_kernel<<<4096, 256, 262144, stream>>>(x, wt, bstd, (float*)d_out);
}
